// MyModel_87522843560680
// MI455X (gfx1250) — matrix operand order repaired and checked by reading, compiled, not run on hardware
//
#include <hip/hip_runtime.h>

typedef __bf16 bf16;
typedef __attribute__((ext_vector_type(16))) __bf16 v16bf;
typedef __attribute__((ext_vector_type(8)))  __bf16 v8bf;
typedef __attribute__((ext_vector_type(8)))  float  v8f;

// global-address-space views so laundered pointers still emit global_load_*
typedef const __attribute__((address_space(1))) bf16*  gcbf;
typedef const __attribute__((address_space(1))) v16bf* gv16p;

#define LL 12
#define HH 512
#define BB 64
#define TT 256
#define FF 512
#define GDIM (4 * HH)     // 2048 gate columns
#define SPLIT 8           // workgroups per layer (each owns 64 h-columns)
#define NTHREADS 512      // 16 waves (wave32): 4 row-tiles x 4 col-quarters

// ---------------------------------------------------------------------------
// One-time conversion kernels (run every launch; deterministic)
// ---------------------------------------------------------------------------

// W,U: [L][H][4H] f32 row-major  ->  Wt,Ut: [L][4H][H] bf16 (n-major / K-contig)
__global__ void cvt_weights(const float* __restrict__ W, const float* __restrict__ U,
                            bf16* __restrict__ Wt, bf16* __restrict__ Ut) {
    size_t i = (size_t)blockIdx.x * blockDim.x + threadIdx.x;
    const size_t total = (size_t)LL * HH * GDIM;
    if (i >= total) return;
    int l = (int)(i / ((size_t)HH * GDIM));
    size_t r = i % ((size_t)HH * GDIM);
    int n = (int)(r / HH);
    int k = (int)(r % HH);
    size_t src = ((size_t)l * HH + k) * GDIM + n;
    Wt[i] = (bf16)W[src];
    Ut[i] = (bf16)U[src];
}

// x: [B][T][F] f32 -> xb: [T][B][F] bf16 (per-step row-major activation)
__global__ void cvt_x(const float* __restrict__ x, bf16* __restrict__ xb) {
    size_t i = (size_t)blockIdx.x * blockDim.x + threadIdx.x;
    const size_t total = (size_t)BB * TT * FF;
    if (i >= total) return;
    int b = (int)(i / ((size_t)TT * FF));
    size_t r = i % ((size_t)TT * FF);
    int t = (int)(r / FF);
    int f = (int)(r % FF);
    xb[((size_t)t * BB + b) * FF + f] = (bf16)x[i];
}

// ---------------------------------------------------------------------------
// Persistent pipelined LSTM
// ---------------------------------------------------------------------------

__device__ __forceinline__ void spin_until(int* p, int target) {
    while (__hip_atomic_load(p, __ATOMIC_ACQUIRE, __HIP_MEMORY_SCOPE_AGENT) < target) {
        __builtin_amdgcn_s_sleep(8);
    }
}

// One [16x512] @ [512 x 64] half-GEMM (4 gate tiles), register ping-pong:
// the 6 loads for iteration ki+1 are issued before the 4 WMMAs of iteration ki.
__device__ __forceinline__ void gemm_half(const bf16* __restrict__ aRow,
                                          const gcbf (&bp)[4],
                                          v8f (&acc)[4], int laneH) {
    const int kA = laneH * 16;
    union AF { v16bf v; v8bf h[2]; };
    AF a[2];
    v16bf bb[2][4];

    a[0].h[0] = *(const v8bf*)(aRow + kA);
    a[0].h[1] = *(const v8bf*)(aRow + kA + 8);
    #pragma unroll
    for (int i = 0; i < 4; ++i) bb[0][i] = *(gv16p)(bp[i]);

    #pragma unroll
    for (int ki = 0; ki < 16; ++ki) {
        const int cur = ki & 1, nxt = cur ^ 1;
        if (ki < 15) {
            const int k0 = (ki + 1) * 32;          // compile-time after unroll
            a[nxt].h[0] = *(const v8bf*)(aRow + k0 + kA);
            a[nxt].h[1] = *(const v8bf*)(aRow + k0 + kA + 8);
            #pragma unroll
            for (int i = 0; i < 4; ++i)
                bb[nxt][i] = *(gv16p)(bp[i] + k0);
        }
        #pragma unroll
        for (int i = 0; i < 4; ++i)
            acc[i] = __builtin_amdgcn_wmma_f32_16x16x32_bf16(
                false, a[cur].v, false, bb[cur][i], (short)0, acc[i], false, false);
    }
}

__global__ void __launch_bounds__(NTHREADS)
lstm_persistent(const bf16* __restrict__ Wt, const bf16* __restrict__ Ut,
                const bf16* __restrict__ xb, const float* __restrict__ bias,
                bf16* __restrict__ hbuf,   // [2][L][B][H] bf16 (parity double buffer)
                float* __restrict__ cbuf,  // [L][B][H] f32 (exclusively owned cells)
                int* __restrict__ flags,   // [L][T] done counters
                float* __restrict__ out)   // [B][H] f32
{
    const int wg   = blockIdx.x;
    const int l    = wg / SPLIT;
    const int sp   = wg % SPLIT;
    const int lane = threadIdx.x & 31;
    const int w    = threadIdx.x >> 5;      // wave 0..15
    const int laneN = lane & 15;
    const int laneH = lane >> 4;            // 0/1
    const int rowBase = (w >> 2) * 16;      // 4 row tiles
    const int colQ    = (w & 3) * 16;       // 4 col quarters per gate
    const int colBase = sp * 64;            // WG's h-column slice

    // --- per-lane constants hoisted out of the time loop ---
    // tile i = gate i (all 4 gates for the same 16 h-columns)
    gcbf wp[4];
    gcbf up[4];
    float bv[4];
    const int hcol = colBase + colQ + laneN;
    #pragma unroll
    for (int g = 0; g < 4; ++g) {
        int n = g * HH + hcol;
        size_t boff = ((size_t)l * GDIM + n) * HH + (size_t)laneH * 16;
        wp[g] = (gcbf)(Wt + boff);
        up[g] = (gcbf)(Ut + boff);
        bv[g] = bias[(size_t)l * GDIM + n];
    }
    const int aRowIdx = rowBase + laneN;    // A-matrix row for this lane
    const size_t layerSlab = (size_t)LL * BB * HH;

    for (int t = 0; t < TT; ++t) {
        // ---- pipeline waits (wavefront over layers) ----
        if (threadIdx.x == 0) {
            if (l > 0)               spin_until(&flags[(l - 1) * TT + t],        SPLIT);
            if (t > 0)               spin_until(&flags[l * TT + (t - 1)],        SPLIT);
            if (l < LL - 1 && t > 1) spin_until(&flags[(l + 1) * TT + (t - 2)],  SPLIT);
        }
        __syncthreads();

        // Launder weight pointers (keeping addrspace(1) provenance) so LICM
        // cannot hoist-and-spill the t-invariant weight loads out of the
        // time loop, while the loads still lower to global_load_b128.
        gcbf wpl[4];
        gcbf upl[4];
        #pragma unroll
        for (int g = 0; g < 4; ++g) {
            gcbf p = wp[g]; asm volatile("" : "+v"(p)); wpl[g] = p;
            gcbf q = up[g]; asm volatile("" : "+v"(q)); upl[g] = q;
        }

        const int par  = t & 1;
        const bf16* inA = (l == 0)
            ? (xb + (size_t)t * BB * FF)
            : (hbuf + (size_t)par * layerSlab + (size_t)(l - 1) * BB * HH);
        const bf16* hA  = hbuf + (size_t)(par ^ 1) * layerSlab + (size_t)l * BB * HH;
        bf16* hOut      = hbuf + (size_t)par * layerSlab + (size_t)l * BB * HH;

        v8f acc[4];
        #pragma unroll
        for (int i = 0; i < 4; ++i)
            acc[i] = (v8f){0.f, 0.f, 0.f, 0.f, 0.f, 0.f, 0.f, 0.f};

        // z = inp @ W + h @ U   (K = 512 + 512, bf16 in / f32 accumulate)
        gemm_half(inA + (size_t)aRowIdx * FF, wpl, acc, laneH);
        gemm_half(hA  + (size_t)aRowIdx * HH, upl, acc, laneH);

        // ---- gate math: each lane holds matching i/f/g/o cells ----
        #pragma unroll
        for (int e = 0; e < 8; ++e) {
            const int row = rowBase + e + 8 * laneH;
            float iv = acc[0][e] + bv[0];
            float fv = acc[1][e] + bv[1];
            float gv = acc[2][e] + bv[2];
            float ov = acc[3][e] + bv[3];
            float ig = 1.f / (1.f + __expf(-iv));
            float fg = 1.f / (1.f + __expf(-fv));
            float gg = tanhf(gv);
            float og = 1.f / (1.f + __expf(-ov));
            size_t cidx = ((size_t)l * BB + row) * HH + hcol;
            float cn = fg * cbuf[cidx] + ig * gg;
            cbuf[cidx] = cn;
            float hn = og * tanhf(cn);
            hOut[(size_t)row * HH + hcol] = (bf16)hn;
            if (l == LL - 1 && t == TT - 1)
                out[(size_t)row * HH + hcol] = hn;
        }

        // ---- publish step completion ----
        __threadfence();
        __syncthreads();
        if (threadIdx.x == 0)
            __hip_atomic_fetch_add(&flags[l * TT + t], 1,
                                   __ATOMIC_RELEASE, __HIP_MEMORY_SCOPE_AGENT);
    }
}

// ---------------------------------------------------------------------------
// Launch
// ---------------------------------------------------------------------------

extern "C" void kernel_launch(void* const* d_in, const int* in_sizes, int n_in,
                              void* d_out, int out_size, void* d_ws, size_t ws_size,
                              hipStream_t stream) {
    const float* x = (const float*)d_in[0];   // [B,T,F]
    const float* W = (const float*)d_in[1];   // [L,H,4H]
    const float* U = (const float*)d_in[2];   // [L,H,4H]
    const float* b = (const float*)d_in[3];   // [L,4H]
    float* out = (float*)d_out;               // [B,H]

    // workspace carve-up (256B aligned slabs)
    const size_t nW   = (size_t)LL * GDIM * HH;          // 12,582,912
    const size_t nX   = (size_t)TT * BB * FF;            //  8,388,608
    const size_t nH2  = (size_t)2 * LL * BB * HH;        //    786,432
    const size_t nC   = (size_t)LL * BB * HH;            //    393,216
    const size_t nFlg = (size_t)LL * TT;                 //      3,072

    char* ws = (char*)d_ws;
    size_t off = 0;
    auto take = [&](size_t bytes) { char* p = ws + off; off = (off + bytes + 255) & ~(size_t)255; return p; };
    bf16* Wt    = (bf16*) take(nW  * sizeof(bf16));
    bf16* Ut    = (bf16*) take(nW  * sizeof(bf16));
    bf16* xb    = (bf16*) take(nX  * sizeof(bf16));
    bf16* hbuf  = (bf16*) take(nH2 * sizeof(bf16));
    float* cbuf = (float*)take(nC  * sizeof(float));
    int*  flags = (int*)  take(nFlg * sizeof(int));
    (void)ws_size; (void)in_sizes; (void)n_in; (void)out_size;

    // reset mutable state every call (graph replays reuse buffers)
    hipMemsetAsync(hbuf,  0, nH2 * sizeof(bf16),  stream);
    hipMemsetAsync(cbuf,  0, nC  * sizeof(float), stream);
    hipMemsetAsync(flags, 0, nFlg * sizeof(int),  stream);

    cvt_weights<<<(unsigned)((nW + 255) / 256), 256, 0, stream>>>(W, U, Wt, Ut);
    cvt_x<<<(unsigned)((nX + 255) / 256), 256, 0, stream>>>(x, xb);

    lstm_persistent<<<LL * SPLIT, NTHREADS, 0, stream>>>(
        Wt, Ut, xb, b, hbuf, cbuf, flags, out);
}
